// GCNSI_41523743817900
// MI455X (gfx1250) — matrix operand order repaired and checked by reading, not compiled, not run on hardware
//
#include <hip/hip_runtime.h>

// ---------------------------------------------------------------------------
// GCNSI on MI455X (gfx1250, wave32).
// (I - alpha*L)^{-1} rhs via Neumann iteration x <- r + (aL)x  (rho(aL)~0.5).
// aL in bf16 (33.5 MB -> L2-resident); matvec = v_wmma_f32_16x16x32_bf16.
// Iterate Xt staged into LDS per-workgroup by the Tensor Data Mover
// (tensor_load_to_lds, TENSORcnt, double-buffered) so 4 waves share it.
// 4096x128x128 GEMM = v_wmma_f32_16x16x4_f32. Message passing = f32 atomics.
// ---------------------------------------------------------------------------

#define N_NODES 4096
#define N_EDGES 131072
#define NITER   24
#define CHUNK   256              // K elements staged per TDM transfer (8KB)
#define NSTAGES (N_NODES / CHUNK)

typedef __attribute__((ext_vector_type(16))) __bf16         v16bf;
typedef __attribute__((ext_vector_type(8)))  float          v8f;
typedef __attribute__((ext_vector_type(2)))  float          v2f;
typedef __attribute__((ext_vector_type(8)))  unsigned short v8u16;
typedef __attribute__((ext_vector_type(4)))  unsigned short v4u16;
typedef __attribute__((ext_vector_type(4)))  unsigned int   u32x4;
typedef __attribute__((ext_vector_type(8)))  int            i32x8;
typedef __attribute__((ext_vector_type(4)))  int            i32x4;

#if __has_builtin(__builtin_amdgcn_tensor_load_to_lds) && \
    __has_builtin(__builtin_amdgcn_s_wait_tensorcnt)
#define HAVE_TDM 1
#else
#define HAVE_TDM 0
#endif

__device__ __forceinline__ unsigned short f2bf(float f) {
    unsigned int u = __float_as_uint(f);
    u += 0x7fffu + ((u >> 16) & 1u);          // round-to-nearest-even
    return (unsigned short)(u >> 16);
}

#if HAVE_TDM
// TDM descriptor for a [16 x CHUNK] bf16 tile of the transposed iterate
// Xt[16][4096] (row = RHS column, contiguous in K). D# layout per ISA 8.3/8.4.
// amdgpu-toolchain (clang-23) 6-arg builtin form.
__device__ __forceinline__ void tdm_load_xt(const unsigned short* gsrc,
                                            unsigned int lds_off) {
    unsigned long long ga = (unsigned long long)(uintptr_t)gsrc;
    u32x4 g0;
    g0[0] = 1u;                                        // count=1, user desc
    g0[1] = lds_off;                                   // lds_addr (bytes)
    g0[2] = (unsigned int)(ga & 0xffffffffull);        // global_addr[31:0]
    g0[3] = (unsigned int)((ga >> 32) & 0x1ffffffull)  // global_addr[56:32]
          | (2u << 30);                                // type=2 ("image")
    i32x8 g1;
    g1[0] = (1 << 16);                                 // data_size=1 -> 2 bytes
    g1[1] = (int)(((unsigned)N_NODES & 0xffffu) << 16);// tensor_dim0[15:0]@63:48
    g1[2] = (int)(((unsigned)N_NODES >> 16)            // tensor_dim0[31:16]
          | (16u << 16));                              // tensor_dim1=16 @95:80
    g1[3] = (int)((unsigned)CHUNK << 16);              // tile_dim0 @127:112
    g1[4] = 16;                                        // tile_dim1=16 @143:128
    g1[5] = N_NODES;                                   // dim0_stride[31:0]
    g1[6] = (int)(((unsigned)N_NODES & 0xffffu) << 16);// dim1_stride[15:0]@223:208
    g1[7] = (int)((unsigned)N_NODES >> 16);            // dim1_stride[47:16]
    i32x4 z4 = { 0, 0, 0, 0 };                         // 2-D: groups 2/3 unused
    i32x8 z8 = { 0, 0, 0, 0, 0, 0, 0, 0 };
    __builtin_amdgcn_tensor_load_to_lds(g0, g1, z4, z4, z8, 0);
}
#endif

// ---------------- prep: degrees, RHS, initial iterate (transposed bf16) -----
__global__ void k_prep1(const float* __restrict__ diff, const float* __restrict__ thrp,
                        float* __restrict__ deg, float* __restrict__ Rp,
                        unsigned short* __restrict__ Xt0) {
    int i = blockIdx.x * blockDim.x + threadIdx.x;
    if (i >= N_NODES) return;
    float thr = thrp[0];
    float d = diff[i];
    float vals[16] = { d, fmaxf(d, thr), fminf(d, thr) };   // rest zero
    deg[i] = 1.0f;                                          // self loop
    #pragma unroll
    for (int c = 0; c < 16; ++c) {
        Rp[i * 16 + c] = vals[c];
        Xt0[(size_t)c * N_NODES + i] = f2bf(vals[c]);       // x0 = r (transposed)
    }
}

__global__ void k_prep2(const int* __restrict__ rows, float* __restrict__ deg) {
    int e = blockIdx.x * blockDim.x + threadIdx.x;
    if (e < N_EDGES) atomicAdd(&deg[rows[e]], 1.0f);
}

__global__ void k_prep3(const float* __restrict__ deg, float* __restrict__ dinv) {
    int i = blockIdx.x * blockDim.x + threadIdx.x;
    if (i < N_NODES) dinv[i] = rsqrtf(deg[i]);
}

// ---------------- M = bf16(alpha * L) ---------------------------------------
__global__ void k_convertM(const float* __restrict__ L, const float* __restrict__ alphap,
                           unsigned short* __restrict__ M) {
    float a = alphap[0];
    size_t idx = ((size_t)blockIdx.x * blockDim.x + threadIdx.x) * 4;
    float4 v = *(const float4*)(L + idx);
    v4u16 o;
    o.x = f2bf(a * v.x); o.y = f2bf(a * v.y); o.z = f2bf(a * v.z); o.w = f2bf(a * v.w);
    *(v4u16*)(M + idx) = o;
}

// ---------------- Neumann sweep: Y = R + M @ X  (bf16 WMMA, fp32 acc) -------
// Block = 4 waves = 64 rows. Xt chunks staged to LDS by TDM (double-buffered),
// shared by all 4 waves; A-matrix streams from global (L2-resident).
__global__ __launch_bounds__(128)
void k_neumann(const unsigned short* __restrict__ M,
               const unsigned short* __restrict__ Xin,   // transposed [16][N]
               unsigned short* __restrict__ Xout,        // transposed [16][N]
               const float* __restrict__ Rp,             // [N][16]
               float* __restrict__ XF,                   // [N][4] features (final)
               const float* __restrict__ diff,
               const float* __restrict__ alphap,
               int final_flag) {
    __shared__ __align__(16) unsigned short xlds[2][16 * CHUNK];   // 2 x 8KB

    // The TDM writes xlds behind the compiler's back: escape the object into
    // an opaque asm with a memory clobber so alias analysis cannot fold the
    // ds_load reads of "never-stored" LDS to undef.
    {
        void* esc = (void*)&xlds[0][0];
        asm volatile("" : "+v"(esc) : : "memory");
    }

    const int lane = threadIdx.x & 31;
    const int wave = threadIdx.x >> 5;
    const int row0 = (blockIdx.x * 4 + wave) * 16;     // 16-row tile
    const int g    = lane >> 4;                        // lane group (ISA layout)
    const int m    = lane & 15;
    const int n    = lane & 15;

    const unsigned short* Arow = M + (size_t)(row0 + m) * N_NODES;

    v8f acc = {};

#if HAVE_TDM
    if (wave == 0) {                                   // TDM ignores EXEC: 1 wave owns it
        tdm_load_xt(Xin, 0u);
        __builtin_amdgcn_s_wait_tensorcnt(0);
        asm volatile("" ::: "memory");                 // LDS now holds stage 0
    }
    __syncthreads();                                   // stage 0 visible to all
#else
    for (int t = threadIdx.x; t < 16 * CHUNK / 8; t += 128) {
        int col = t / (CHUNK / 8), kq = t % (CHUNK / 8);
        ((v8u16*)&xlds[0][0])[t] =
            *(const v8u16*)(Xin + (size_t)col * N_NODES + kq * 8);
    }
    __syncthreads();
#endif

    for (int s = 0; s < NSTAGES; ++s) {
        const int cur = s & 1;
#if HAVE_TDM
        if (wave == 0 && (s + 1) < NSTAGES)            // prefetch next chunk
            tdm_load_xt(Xin + (size_t)(s + 1) * CHUNK,
                        (unsigned int)(((s + 1) & 1) * (16 * CHUNK * 2)));
#endif
        const unsigned short* Bl = &xlds[cur][0];
        const int kg = s * CHUNK;
        #pragma unroll
        for (int kk = 0; kk < CHUNK; kk += 32) {
            union { v8u16 h[2]; v16bf v; } a, b;
            // A 16x32 bf16 frag: elems 0..7 -> K=8g+e ; elems 8..15 -> K=16+8g+e
            a.h[0] = *(const v8u16*)(Arow + kg + kk + 8 * g);
            a.h[1] = *(const v8u16*)(Arow + kg + kk + 16 + 8 * g);
            // B 32x16 bf16 frag from LDS: elems 0..7 -> K=8g+e ; elems 8..15 -> K=16+8g+e, col n
            b.h[0] = *(const v8u16*)(Bl + n * CHUNK + kk + 8 * g);
            b.h[1] = *(const v8u16*)(Bl + n * CHUNK + kk + 16 + 8 * g);
            __builtin_prefetch(Arow + kg + kk + 512, 0, 1);   // global_prefetch_b8
            acc = __builtin_amdgcn_wmma_f32_16x16x32_bf16(
                      false, a.v, false, b.v, (short)0, acc, false, false);
        }
        __syncthreads();                               // all waves done with buf cur
#if HAVE_TDM
        if (wave == 0 && (s + 1) < NSTAGES) {
            __builtin_amdgcn_s_wait_tensorcnt(0);      // next chunk landed
            asm volatile("" ::: "memory");
        }
#else
        if ((s + 1) < NSTAGES)
            for (int t = threadIdx.x; t < 16 * CHUNK / 8; t += 128) {
                int col = t / (CHUNK / 8), kq = t % (CHUNK / 8);
                ((v8u16*)&xlds[(s + 1) & 1][0])[t] =
                    *(const v8u16*)(Xin + (size_t)col * N_NODES + (s + 1) * CHUNK + kq * 8);
            }
#endif
        __syncthreads();                               // hand buffer to consumers
    }

    if (!final_flag) {
        #pragma unroll
        for (int v = 0; v < 8; ++v) {                  // C/D: M = v+8g, N = n
            int r = row0 + v + 8 * g;
            float y = Rp[r * 16 + n] + acc[v];
            Xout[(size_t)n * N_NODES + r] = f2bf(y);
        }
    } else {
        float oma = 1.0f - alphap[0];
        #pragma unroll
        for (int v = 0; v < 8; ++v) {
            int r = row0 + v + 8 * g;
            float y = Rp[r * 16 + n] + acc[v];
            if (n < 3)  XF[r * 4 + 1 + n] = oma * y;   // sol columns
            if (n == 3) XF[r * 4 + 0]     = diff[r];   // feature col 0
        }
    }
}

// ---------------- tiny GEMM: B1 = XF[N,4] @ w1[4,128] + b1 ------------------
__global__ void k_gemm_xw1(const float* __restrict__ XF, const float* __restrict__ w1,
                           const float* __restrict__ b1, float* __restrict__ O) {
    int t = blockIdx.x * blockDim.x + threadIdx.x;
    if (t >= N_NODES * 128) return;
    int i = t >> 7, j = t & 127;
    float s = b1[j];
    #pragma unroll
    for (int k = 0; k < 4; ++k) s += XF[i * 4 + k] * w1[k * 128 + j];
    O[t] = s;
}

// ---------------- g = dinv*buf ; agg = g (self loop) ------------------------
__global__ void k_scale_init(float* __restrict__ buf, float* __restrict__ agg,
                             const float* __restrict__ dinv) {
    int t = blockIdx.x * blockDim.x + threadIdx.x;
    if (t >= N_NODES * 128) return;
    float v = dinv[t >> 7] * buf[t];
    buf[t] = v;
    agg[t] = v;
}

// ---------------- edge-parallel scatter: agg[col] += g[row] -----------------
__global__ void k_scatter(const float* __restrict__ g, float* __restrict__ agg,
                          const int* __restrict__ rows, const int* __restrict__ cols) {
    int t = blockIdx.x * blockDim.x + threadIdx.x;
    int e = t >> 5;
    if (e >= N_EDGES) return;
    int c4 = (t & 31) * 4;
    int r = rows[e], c = cols[e];
    float4 v = *(const float4*)(g + (size_t)r * 128 + c4);
    float* dst = agg + (size_t)c * 128 + c4;
    atomicAdd(dst + 0, v.x); atomicAdd(dst + 1, v.y);
    atomicAdd(dst + 2, v.z); atomicAdd(dst + 3, v.w);
}

// ---------------- dst = act(dinv * agg) -------------------------------------
__global__ void k_finish(const float* __restrict__ agg, float* __restrict__ dst,
                         const float* __restrict__ dinv, int do_relu) {
    int t = blockIdx.x * blockDim.x + threadIdx.x;
    if (t >= N_NODES * 128) return;
    float v = dinv[t >> 7] * agg[t];
    if (do_relu) v = fmaxf(v, 0.0f);
    dst[t] = v;
}

// ---------------- O = A[N,128] @ W[128,128] + bias  (fp32 WMMA 16x16x4) -----
__global__ __launch_bounds__(128)
void k_gemm128_wmma(const float* __restrict__ A, const float* __restrict__ W,
                    const float* __restrict__ bias, float* __restrict__ O) {
    const int lane = threadIdx.x & 31;
    const int wave = threadIdx.x >> 5;
    const int row0 = (blockIdx.x * 4 + wave) * 16;
    const int g    = lane >> 4;
    const int m    = lane & 15;
    const int n    = lane & 15;

    v8f acc[8];
    #pragma unroll
    for (int t = 0; t < 8; ++t) acc[t] = (v8f){};

    for (int k = 0; k < 128; k += 4) {
        // A 16x4 f32 frag: elem e -> K = k + 2g + e (contiguous pair)
        v2f a = *(const v2f*)(A + (size_t)(row0 + m) * 128 + k + 2 * g);
        #pragma unroll
        for (int t = 0; t < 8; ++t) {                  // 8 N-tiles share A frag
            v2f b;
            b[0] = W[(k + 2 * g) * 128 + t * 16 + n];  // B 4x16: elem e -> K=k+2g+e
            b[1] = W[(k + 2 * g + 1) * 128 + t * 16 + n];
            acc[t] = __builtin_amdgcn_wmma_f32_16x16x4_f32(
                         false, a, false, b, (short)0, acc[t], false, false);
        }
    }
    #pragma unroll
    for (int t = 0; t < 8; ++t)
        #pragma unroll
        for (int v = 0; v < 8; ++v) {
            int r = row0 + v + 8 * g;
            int c = t * 16 + n;
            O[(size_t)r * 128 + c] = acc[t][v] + bias[c];
        }
}

// ---------------- head: out[N,2] = (dinv*agg) @ wf + bf ---------------------
__global__ void k_head(const float* __restrict__ agg, const float* __restrict__ dinv,
                       const float* __restrict__ wf, const float* __restrict__ bfv,
                       float* __restrict__ out) {
    int t = blockIdx.x * blockDim.x + threadIdx.x;
    if (t >= N_NODES * 2) return;
    int i = t >> 1, o = t & 1;
    float di = dinv[i], s = bfv[o];
    #pragma unroll 8
    for (int j = 0; j < 128; ++j) s += di * agg[i * 128 + j] * wf[j * 2 + o];
    out[t] = s;
}

// ---------------------------------------------------------------------------
extern "C" void kernel_launch(void* const* d_in, const int* in_sizes, int n_in,
                              void* d_out, int out_size, void* d_ws, size_t ws_size,
                              hipStream_t stream) {
    const float* alphap = (const float*)d_in[0];
    const float* L      = (const float*)d_in[1];
    const float* thrp   = (const float*)d_in[2];
    const float* diff   = (const float*)d_in[3];
    const int*   ei     = (const int*)d_in[4];
    const float* w1     = (const float*)d_in[5];
    const float* b1     = (const float*)d_in[6];
    const float* w2     = (const float*)d_in[7];
    const float* b2     = (const float*)d_in[8];
    const float* wf     = (const float*)d_in[9];
    const float* bfv    = (const float*)d_in[10];
    float*       out    = (float*)d_out;
    const int* rows = ei;
    const int* cols = ei + N_EDGES;

    // workspace carve-up (bytes)
    unsigned char* ws = (unsigned char*)d_ws;
    unsigned short* M    = (unsigned short*)(ws);                      // 33,554,432
    unsigned short* XT0  = (unsigned short*)(ws + 33554432);           //    131,072
    unsigned short* XT1  = (unsigned short*)(ws + 33685504);           //    131,072
    float*          RP   = (float*)(ws + 33816576);                    //    262,144
    float*          DEG  = (float*)(ws + 34078720);                    //     16,384
    float*          DINV = (float*)(ws + 34095104);                    //     16,384
    float*          XF   = (float*)(ws + 34111488);                    //     65,536
    float*          B1   = (float*)(ws + 34177024);                    //  2,097,152
    float*          B2   = (float*)(ws + 36274176);                    //  2,097,152

    // --- prep ---
    k_prep1<<<(N_NODES + 255) / 256, 256, 0, stream>>>(diff, thrp, DEG, RP, XT0);
    k_prep2<<<(N_EDGES + 255) / 256, 256, 0, stream>>>(rows, DEG);
    k_prep3<<<(N_NODES + 255) / 256, 256, 0, stream>>>(DEG, DINV);
    k_convertM<<<16384, 256, 0, stream>>>(L, alphap, M);

    // --- Neumann sweeps (bf16 WMMA, TDM-staged iterate, L2-resident matrix) ---
    unsigned short* xin = XT0; unsigned short* xout = XT1;
    for (int it = 0; it < NITER; ++it) {
        k_neumann<<<64, 128, 0, stream>>>(M, xin, xout, RP, XF, diff, alphap, 0);
        unsigned short* t = xin; xin = xout; xout = t;
    }
    k_neumann<<<64, 128, 0, stream>>>(M, xin, xout, RP, XF, diff, alphap, 1);

    // --- GCN conv 1 ---
    k_gemm_xw1<<<(N_NODES * 128) / 256, 256, 0, stream>>>(XF, w1, b1, B1);
    k_scale_init<<<(N_NODES * 128) / 256, 256, 0, stream>>>(B1, B2, DINV);
    k_scatter<<<(N_EDGES * 32) / 256, 256, 0, stream>>>(B1, B2, rows, cols);
    k_finish<<<(N_NODES * 128) / 256, 256, 0, stream>>>(B2, B1, DINV, 1); // relu

    // --- GCN conv 2 (fp32 WMMA GEMM) ---
    k_gemm128_wmma<<<64, 128, 0, stream>>>(B1, w2, b2, B2);
    k_scale_init<<<(N_NODES * 128) / 256, 256, 0, stream>>>(B2, B1, DINV);
    k_scatter<<<(N_EDGES * 32) / 256, 256, 0, stream>>>(B2, B1, rows, cols);

    // --- head ---
    k_head<<<(N_NODES * 2 + 255) / 256, 256, 0, stream>>>(B1, DINV, wf, bfv, out);
}
